// SelfAttentionV3_43130061586977
// MI455X (gfx1250) — matrix operand order repaired and checked by reading, not compiled, not run on hardware
//
#include <hip/hip_runtime.h>

// ---------------- types ----------------
typedef __attribute__((ext_vector_type(16))) __bf16 v16bf;
typedef __attribute__((ext_vector_type(8)))  float  v8f;

union FragBf16 {
    v16bf v;
    uint4 u[2];
};

#define BM 128
#define BN 256
#define BK 32
#define LDSA 40   // bf16 elements per LDS row (32 + 8 pad, keeps 16B alignment)
#define LDSB 40

// =====================================================================
// Generic NT GEMM:  C[m,n] = scale * sum_k A[m,k]*B[n,k] + bias[n]
// A: [M,K] bf16, row stride lda;  B: [N,K] bf16, row stride ldb.
// Writes fp32 (Cf) and/or bf16 (Ch). blockIdx.z batches via strides.
// Block tile 128x256, 8 waves as 2(M) x 4(N), wave tile 64x64 = 4x4
// sub-tiles of v_wmma_f32_16x16x32_bf16. 16 ds_load_b128 per 16 WMMA.
// Requires M%128==0, N%256==0, K%32==0 (true for all stages here).
// =====================================================================
__global__ __launch_bounds__(256)
void gemm_nt_bf16(const __bf16* __restrict__ A, int lda, long strideAz,
                  const __bf16* __restrict__ Bp, int ldb, long strideBz,
                  float* __restrict__ Cf, __bf16* __restrict__ Ch,
                  int ldc, long strideCz,
                  const float* __restrict__ bias,
                  int K, float scale)
{
    __shared__ __bf16 As[BM * LDSA];
    __shared__ __bf16 Bs[BN * LDSB];

    const int z = blockIdx.z;
    A  += (long)z * strideAz;
    Bp += (long)z * strideBz;

    const int tid  = threadIdx.x;
    const int lane = tid & 31;
    const int wave = tid >> 5;      // 0..7
    const int wm   = wave >> 2;     // 0..1 : 64 rows of M each
    const int wn   = wave & 3;      // 0..3 : 64 cols of N each

    const int blockM = blockIdx.x * BM;
    const int blockN = blockIdx.y * BN;

    const int h  = lane >> 4;       // half-lane group (0/1)
    const int ml = lane & 15;       // m (A/C rows) or n (B/C cols) in tile

    // staging: 256 threads; A: 2 chunks of 8 bf16, B: 4 chunks of 8 bf16
    const int srow = tid >> 2;          // 0..63
    const int scol = (tid & 3) * 8;     // 0,8,16,24 (bf16 elements)

    v8f acc[4][4];
#pragma unroll
    for (int i = 0; i < 4; ++i)
#pragma unroll
        for (int j = 0; j < 4; ++j)
            acc[i][j] = (v8f){0.f,0.f,0.f,0.f,0.f,0.f,0.f,0.f};

    for (int k0 = 0; k0 < K; k0 += BK) {
        // ---- stage A tile (128 x 32) and B tile (256 x 32) ----
        {
            const __bf16* ga0 = A + (long)(blockM + srow)      * lda + k0 + scol;
            const __bf16* ga1 = A + (long)(blockM + srow + 64) * lda + k0 + scol;
            *(uint4*)(&As[(srow     ) * LDSA + scol]) = *(const uint4*)ga0;
            *(uint4*)(&As[(srow + 64) * LDSA + scol]) = *(const uint4*)ga1;
#pragma unroll
            for (int p = 0; p < 4; ++p) {
                const __bf16* gb = Bp + (long)(blockN + srow + 64 * p) * ldb + k0 + scol;
                *(uint4*)(&Bs[(srow + 64 * p) * LDSB + scol]) = *(const uint4*)gb;
            }
        }
        __syncthreads();

        // ---- B fragments: lane ml = n; elements 0..7 K = 8h..8h+7, 8..15 K = 16+8h..16+8h+7 ----
        FragBf16 fb[4];
#pragma unroll
        for (int j = 0; j < 4; ++j) {
            const int row = wn * 64 + j * 16 + ml;
            const __bf16* p = &Bs[row * LDSB + h * 8];
            fb[j].u[0] = *(const uint4*)(p);
            fb[j].u[1] = *(const uint4*)(p + 16);
        }

        // ---- A fragments + WMMA (4 x 4) ----
#pragma unroll
        for (int i = 0; i < 4; ++i) {
            const int row = wm * 64 + i * 16 + ml;
            FragBf16 fa;
            const __bf16* p = &As[row * LDSA + h * 8];
            fa.u[0] = *(const uint4*)(p);        // K = h*8 .. h*8+7
            fa.u[1] = *(const uint4*)(p + 16);   // K = 16+h*8 .. 16+h*8+7
#pragma unroll
            for (int j = 0; j < 4; ++j) {
                acc[i][j] = __builtin_amdgcn_wmma_f32_16x16x32_bf16(
                    false, fa.v, false, fb[j].v, (short)0, acc[i][j],
                    false, false);
            }
        }
        __syncthreads();
    }

    // ---- epilogue: C layout m = r + 8h, n = ml ----
#pragma unroll
    for (int i = 0; i < 4; ++i) {
        const int gmBase = blockM + wm * 64 + i * 16 + h * 8;
#pragma unroll
        for (int j = 0; j < 4; ++j) {
            const int gn = blockN + wn * 64 + j * 16 + ml;
            const float bv = bias ? bias[gn] : 0.0f;
#pragma unroll
            for (int r = 0; r < 8; ++r) {
                const long idx = (long)z * strideCz + (long)(gmBase + r) * ldc + gn;
                const float v = acc[i][j][r] * scale + bv;
                if (Cf) Cf[idx] = v;
                if (Ch) Ch[idx] = (__bf16)v;
            }
        }
    }
}

// =====================================================================
// float -> bf16 convert (grid-stride)
// =====================================================================
__global__ __launch_bounds__(256)
void f32_to_bf16(const float* __restrict__ in, __bf16* __restrict__ out, long n)
{
    long i = (long)blockIdx.x * 256 + threadIdx.x;
    const long stride = (long)gridDim.x * 256;
    for (; i < n; i += stride)
        out[i] = (__bf16)in[i];
}

// =====================================================================
// Row softmax, in place on bf16 scores. One block per row (2048 cols).
// fp32 math, mask==0 -> -inf (reference semantics).
// =====================================================================
__global__ __launch_bounds__(256)
void softmax_rows(__bf16* __restrict__ S, const int* __restrict__ mask, int ncols)
{
    const long row = blockIdx.x;
    __bf16* p = S + row * (long)ncols;
    const int* mrow = mask + row * (long)ncols;
    const int tid = threadIdx.x;

    float vals[8];
    float mx = -INFINITY;
#pragma unroll
    for (int i = 0; i < 8; ++i) {
        const int c = tid + i * 256;
        float v = (float)p[c];
        if (mrow[c] == 0) v = -INFINITY;
        vals[i] = v;
        mx = fmaxf(mx, v);
    }
#pragma unroll
    for (int off = 16; off > 0; off >>= 1)
        mx = fmaxf(mx, __shfl_xor(mx, off, 32));

    __shared__ float redMax[8];
    __shared__ float redSum[8];
    if ((tid & 31) == 0) redMax[tid >> 5] = mx;
    __syncthreads();
    mx = redMax[0];
#pragma unroll
    for (int w = 1; w < 8; ++w) mx = fmaxf(mx, redMax[w]);

    float sum = 0.f;
#pragma unroll
    for (int i = 0; i < 8; ++i) {
        vals[i] = __expf(vals[i] - mx);
        sum += vals[i];
    }
#pragma unroll
    for (int off = 16; off > 0; off >>= 1)
        sum += __shfl_xor(sum, off, 32);
    if ((tid & 31) == 0) redSum[tid >> 5] = sum;
    __syncthreads();
    sum = 0.f;
#pragma unroll
    for (int w = 0; w < 8; ++w) sum += redSum[w];

    const float inv = 1.0f / sum;
#pragma unroll
    for (int i = 0; i < 8; ++i)
        p[tid + i * 256] = (__bf16)(vals[i] * inv);
}

// =====================================================================
// Transpose V slice of QKV ([B*S, 3072] cols 2048..3071) into Vt [B,E,S]
// so that ctx GEMM is NT (K-contiguous on both sides).
// =====================================================================
__global__ __launch_bounds__(256)
void transpose_v(const __bf16* __restrict__ Vsrc, __bf16* __restrict__ Vt)
{
    __shared__ __bf16 tile[32][33];
    const int b  = blockIdx.z;
    const int s0 = blockIdx.x * 32;
    const int e0 = blockIdx.y * 32;
    const int tx = threadIdx.x & 31;
    const int ty = threadIdx.x >> 5;   // 0..7
#pragma unroll
    for (int i = 0; i < 32; i += 8)
        tile[ty + i][tx] = Vsrc[(long)(b * 2048 + s0 + ty + i) * 3072 + e0 + tx];
    __syncthreads();
#pragma unroll
    for (int i = 0; i < 32; i += 8)
        Vt[(long)b * 1024 * 2048 + (long)(e0 + ty + i) * 2048 + s0 + tx] = tile[tx][ty + i];
}

// =====================================================================
// Host-side launch
// =====================================================================
extern "C" void kernel_launch(void* const* d_in, const int* in_sizes, int n_in,
                              void* d_out, int out_size, void* d_ws, size_t ws_size,
                              hipStream_t stream)
{
    const float* X     = (const float*)d_in[0];  // [4,2048,1024]
    const int*   mask  = (const int*)  d_in[1];  // [4,2048,2048]
    const float* Wqkv  = (const float*)d_in[2];  // [3072,1024]
    const float* bqkv  = (const float*)d_in[3];  // [3072]
    const float* Wout  = (const float*)d_in[4];  // [1024,1024]
    const float* bout  = (const float*)d_in[5];  // [1024]
    float* out = (float*)d_out;                  // [4,2048,1024] fp32

    const int  Bn = 4, S = 2048, E = 1024;
    const long BS = (long)Bn * S;                // 8192

    char* ws = (char*)d_ws;
    size_t off = 0;
    auto alloc = [&](size_t bytes) -> char* {
        char* p = ws + off;
        off += (bytes + 255) & ~(size_t)255;
        return p;
    };

    __bf16* Xb    = (__bf16*)alloc(BS * E * 2);             // 16 MB
    __bf16* Wqkvb = (__bf16*)alloc((size_t)3 * E * E * 2);  // 6 MB
    __bf16* Woutb = (__bf16*)alloc((size_t)E * E * 2);      // 2 MB
    __bf16* QKVb  = (__bf16*)alloc(BS * 3 * E * 2);         // 48 MB
    __bf16* Sb    = (__bf16*)alloc((size_t)Bn * S * S * 2); // 32 MB
    __bf16* Vtb   = (__bf16*)alloc((size_t)Bn * E * S * 2); // 16 MB
    __bf16* Ctxb  = (__bf16*)alloc(BS * E * 2);             // 16 MB

    // 1) fp32 -> bf16 converts
    f32_to_bf16<<<2048, 256, 0, stream>>>(X,    Xb,    BS * E);
    f32_to_bf16<<<1024, 256, 0, stream>>>(Wqkv, Wqkvb, (long)3 * E * E);
    f32_to_bf16<<<512,  256, 0, stream>>>(Wout, Woutb, (long)E * E);

    // 2) QKV = X * Wqkv^T + b_qkv   [8192,3072], K=1024
    {
        dim3 grid(BS / BM, (3 * E) / BN, 1);
        gemm_nt_bf16<<<grid, 256, 0, stream>>>(
            Xb, E, 0, Wqkvb, E, 0,
            nullptr, QKVb, 3 * E, 0,
            bqkv, E, 1.0f);
    }

    // 3) Vt[b,e,s] = V[b,s,e]
    {
        dim3 grid(S / 32, E / 32, Bn);
        transpose_v<<<grid, 256, 0, stream>>>(QKVb + 2 * E, Vtb);
    }

    // 4) S = Q * K^T / sqrt(E)   per batch [2048,2048], K=1024
    {
        dim3 grid(S / BM, S / BN, Bn);
        gemm_nt_bf16<<<grid, 256, 0, stream>>>(
            QKVb,     3 * E, (long)S * 3 * E,     // Q slice
            QKVb + E, 3 * E, (long)S * 3 * E,     // K slice
            nullptr, Sb, S, (long)S * S,
            nullptr, E, 0.03125f);                // 1/sqrt(1024)
    }

    // 5) softmax rows (in place, applies mask)
    softmax_rows<<<(unsigned)BS, 256, 0, stream>>>(Sb, mask, S);

    // 6) ctx = P * V   per batch [2048,1024], K=2048 (B operand = Vt, NT)
    {
        dim3 grid(S / BM, E / BN, Bn);
        gemm_nt_bf16<<<grid, 256, 0, stream>>>(
            Sb,  S, (long)S * S,
            Vtb, S, (long)E * S,
            nullptr, Ctxb, E, (long)S * E,
            nullptr, S, 1.0f);
    }

    // 7) out = ctx * Wout^T + b_out   [8192,1024], K=1024, fp32 output
    {
        dim3 grid(BS / BM, E / BN, 1);
        gemm_nt_bf16<<<grid, 256, 0, stream>>>(
            Ctxb, E, 0, Woutb, E, 0,
            out, nullptr, E, 0,
            bout, E, 1.0f);
    }
}
